// MemoAI_2491081032431
// MI455X (gfx1250) — compile-verified, matrix operand order checked by reading, not run on hardware
//
#include <hip/hip_runtime.h>
#include <cstdint>

// ---------------- problem dims (compile-time constants) ----------------
#define BB  2
#define SS  1024
#define DD  1024
#define HH  16
#define FF  2048
#define EE  8
#define HDD 64
#define NT  (BB*SS)   // 2048 tokens

typedef __bf16 bf16;
typedef __bf16 v16bf __attribute__((ext_vector_type(16)));
typedef __bf16 v8bf  __attribute__((ext_vector_type(8)));
typedef float  v8f   __attribute__((ext_vector_type(8)));

__device__ __forceinline__ float sigmoidf_(float x) { return 1.0f / (1.0f + expf(-x)); }

// =======================================================================
// Generic strided/batched/gathered bf16 WMMA GEMM:
//   C[z][m][n] = sum_k A[z][m][k] * Bt[z][n][k]   (Bt is N-major, K-contig)
// epilogue: (+bias[n]) (*rowscale[m]) -> f32 and/or bf16 stores.
// Block = 128 threads (4 waves), tile 128x64, K-step 32, double-buffered
// LDS with CDNA5 async global->LDS staging for the B operand.
// Each wave owns a 32x64 strip: 2 A-frags x 4 B-frags = 8 back-to-back WMMAs.
// =======================================================================
struct GemmParams {
  const bf16* A; const bf16* Bt;
  float* Cf; bf16* Cb;
  const float* bias; const float* rowscale;
  const int* rowidx; const int* counts;     // grouped-MoE gather / per-batch M
  long lda, ldb, ldc;
  long bsA1, bsA2, bsB1, bsB2, bsC1, bsC2;  // batch offsets: zq*bs1 + zr*bs2
  long bsBias, bsRS1, bsRS2;
  int  zdiv, M, N, K, rowidxStride;
};

#define BKP 40   // 32 + 8 pad (row stride 80 B keeps 16B alignment)
#define BM  128

__global__ void __launch_bounds__(128) gemm_bf16_wmma(GemmParams p) {
  __shared__ __align__(16) bf16 As[2][BM][BKP];   // 20.0 KB
  __shared__ __align__(16) bf16 Bs[2][64][BKP];   // 10.0 KB

  const int z  = blockIdx.z;
  const int zq = z / p.zdiv, zr = z % p.zdiv;
  const bf16* Ab = p.A  + (long)zq * p.bsA1 + (long)zr * p.bsA2;
  const bf16* Bb = p.Bt + (long)zq * p.bsB1 + (long)zr * p.bsB2;
  const long cOff = (long)zq * p.bsC1 + (long)zr * p.bsC2;
  const int mbase = blockIdx.y * BM, nbase = blockIdx.x * 64;
  const int mcount = p.counts ? p.counts[z] : p.M;
  if (mbase >= mcount) return;
  const int* ridx = p.rowidx ? (p.rowidx + (long)z * p.rowidxStride) : nullptr;

  const int tid = threadIdx.x, lane = tid & 31, warp = tid >> 5;

  // ---- loop-invariant staging addresses (gather hoisted out of k-loop) ----
  const int  arow   = mbase + tid;                 // one A row per thread
  const bool avalid = arow < mcount;
  long aoff = 0;
  if (avalid) aoff = (ridx ? (long)ridx[arow] : (long)arow) * p.lda;
  const bf16* aRow = Ab + aoff;

  const bf16* bRow = Bb + (long)(nbase + (tid >> 1)) * p.ldb + (tid & 1) * 16;
  const unsigned bLds0 = (unsigned)(uintptr_t)&Bs[0][tid >> 1][(tid & 1) * 16];
  const unsigned bLds1 = (unsigned)(uintptr_t)&Bs[1][tid >> 1][(tid & 1) * 16];
  const uint4 z4 = make_uint4(0, 0, 0, 0);

#define STAGE_TILES(bufi, kk)                                                  \
  {                                                                            \
    uint4 d0 = z4, d1 = z4, d2 = z4, d3 = z4;                                  \
    if (avalid) {                                                              \
      const uint4* g = (const uint4*)(aRow + (kk));                            \
      d0 = g[0]; d1 = g[1]; d2 = g[2]; d3 = g[3];                              \
    }                                                                          \
    uint4* dst = (uint4*)&As[bufi][tid][0];                                    \
    dst[0] = d0; dst[1] = d1; dst[2] = d2; dst[3] = d3;                        \
    unsigned lo = (bufi) ? bLds1 : bLds0;                                      \
    unsigned long long ga = (unsigned long long)(uintptr_t)(bRow + (kk));      \
    asm volatile("global_load_async_to_lds_b128 %0, %1, off"                   \
                 :: "v"(lo), "v"(ga) : "memory");                              \
    asm volatile("global_load_async_to_lds_b128 %0, %1, off"                   \
                 :: "v"(lo + 16u), "v"(ga + 16ull) : "memory");                \
  }

  v8f acc[8];
  const v8f vzero = {0.f,0.f,0.f,0.f,0.f,0.f,0.f,0.f};
#pragma unroll
  for (int t = 0; t < 8; ++t) acc[t] = vzero;

  const int kA = (lane < 16) ? 0 : 8;        // ISA 7.12.2 16-bit A/B frag layout
  const int mr = warp * 32 + (lane & 15);
  const int nr = lane & 15;

  const int nk = p.K >> 5;

  // prologue: stage tile 0 into buffer 0
  STAGE_TILES(0, 0)
  asm volatile("s_wait_asynccnt 0" ::: "memory");
  __syncthreads();

  for (int i = 0; i < nk; ++i) {
    const int cur = i & 1;
    if (i + 1 < nk) {
      if (cur) STAGE_TILES(0, (i + 1) << 5)
      else     STAGE_TILES(1, (i + 1) << 5)
    }

    // ---- load all fragments for this tile, then 8 back-to-back WMMAs ----
    v8bf a0lo = *(const v8bf*)&As[cur][mr][kA];
    v8bf a0hi = *(const v8bf*)&As[cur][mr][kA + 16];
    v8bf a1lo = *(const v8bf*)&As[cur][mr + 16][kA];
    v8bf a1hi = *(const v8bf*)&As[cur][mr + 16][kA + 16];
    v16bf a0 = __builtin_shufflevector(a0lo, a0hi, 0,1,2,3,4,5,6,7,8,9,10,11,12,13,14,15);
    v16bf a1 = __builtin_shufflevector(a1lo, a1hi, 0,1,2,3,4,5,6,7,8,9,10,11,12,13,14,15);

    v16bf bfr[4];
#pragma unroll
    for (int t = 0; t < 4; ++t) {
      v8bf blo = *(const v8bf*)&Bs[cur][t * 16 + nr][kA];
      v8bf bhi = *(const v8bf*)&Bs[cur][t * 16 + nr][kA + 16];
      bfr[t] = __builtin_shufflevector(blo, bhi, 0,1,2,3,4,5,6,7,8,9,10,11,12,13,14,15);
    }
#pragma unroll
    for (int t = 0; t < 4; ++t) {
      acc[t]     = __builtin_amdgcn_wmma_f32_16x16x32_bf16(
          false, a0, false, bfr[t], (short)0, acc[t],     false, false);
      acc[4 + t] = __builtin_amdgcn_wmma_f32_16x16x32_bf16(
          false, a1, false, bfr[t], (short)0, acc[4 + t], false, false);
    }

    asm volatile("s_wait_asynccnt 0" ::: "memory");  // our async stores landed
    __syncthreads();                                  // all waves done read+write
  }
#undef STAGE_TILES

  // ---- epilogue (C layout: lanes 0-15 -> M=v, lanes 16-31 -> M=v+8) ----
#pragma unroll
  for (int s = 0; s < 2; ++s) {
#pragma unroll
    for (int t = 0; t < 4; ++t) {
#pragma unroll
      for (int v = 0; v < 8; ++v) {
        int r = mbase + warp * 32 + s * 16 + v + ((lane & 16) ? 8 : 0);
        if (r < mcount) {
          long rg = ridx ? (long)ridx[r] : (long)r;
          int c = nbase + t * 16 + (lane & 15);
          float val = acc[s * 4 + t][v];
          if (p.bias)     val += p.bias[(long)z * p.bsBias + c];
          if (p.rowscale) val *= p.rowscale[(long)zq * p.bsRS1 + (long)zr * p.bsRS2 + r];
          long o = cOff + rg * p.ldc + c;
          if (p.Cf) p.Cf[o] = val;
          if (p.Cb) p.Cb[o] = (bf16)val;
        }
      }
    }
  }
}

// =======================================================================
// Small helper kernels
// =======================================================================
__global__ void cvt_f32_bf16(const float* in, bf16* out, long n) {
  long i = (long)blockIdx.x * blockDim.x + threadIdx.x;
  if (i < n) out[i] = (bf16)in[i];
}

// transpose-convert: in [batch][R][C] f32 -> out [batch][C][R] bf16
__global__ void tconv_kernel(const float* in, bf16* out, int R, int C) {
  long b = blockIdx.z;
  const float* ip = in + b * (long)R * C;
  bf16* op = out + b * (long)R * C;
  int c = blockIdx.x * 16 + threadIdx.x;
  int r = blockIdx.y * 16 + threadIdx.y;
  if (r < R && c < C) op[(long)c * R + r] = (bf16)ip[(long)r * C + c];
}

// V [token][D] bf16 -> Vt [b][h][hd][s] bf16
__global__ void vt_kernel(const bf16* Vb, bf16* Vt) {
  long i = (long)blockIdx.x * blockDim.x + threadIdx.x;
  if (i >= (long)BB * HH * HDD * SS) return;
  int s = (int)(i % SS); long r = i / SS;
  int hd = (int)(r % HDD); r /= HDD;
  int h = (int)(r % HH); int b = (int)(r / HH);
  Vt[i] = Vb[((long)(b * SS + s)) * DD + h * HDD + hd];
}

// rowscale[b][h][s] = sigmoid(q . scale_w[h]) * 2 / sqrt(64) = sigmoid(.)/4
__global__ void scale_kernel(const float* Qf, const float* sw, float* sc) {
  int i = blockIdx.x * blockDim.x + threadIdx.x;
  if (i >= BB * HH * SS) return;
  int s = i % SS; int h = (i / SS) % HH; int b = i / (SS * HH);
  const float* q = Qf + ((long)(b * SS + s)) * DD + h * HDD;
  const float* w = sw + h * HDD;
  float d = 0.f;
#pragma unroll 8
  for (int k = 0; k < HDD; ++k) d += q[k] * w[k];
  sc[i] = sigmoidf_(d) * 0.25f;
}

// in-place row softmax over 1024 bf16 (scores already scaled)
__global__ void __launch_bounds__(256) softmax_kernel(bf16* sc) {
  __shared__ float red[256];
  bf16* p = sc + (long)blockIdx.x * SS;
  int t = threadIdx.x;
  float v[4]; float mx = -3.0e38f;
#pragma unroll
  for (int i = 0; i < 4; ++i) { v[i] = (float)p[t + i * 256]; mx = fmaxf(mx, v[i]); }
  red[t] = mx; __syncthreads();
  for (int s = 128; s > 0; s >>= 1) { if (t < s) red[t] = fmaxf(red[t], red[t + s]); __syncthreads(); }
  mx = red[0]; __syncthreads();
  float sum = 0.f;
#pragma unroll
  for (int i = 0; i < 4; ++i) { v[i] = expf(v[i] - mx); sum += v[i]; }
  red[t] = sum; __syncthreads();
  for (int s = 128; s > 0; s >>= 1) { if (t < s) red[t] += red[t + s]; __syncthreads(); }
  float inv = 1.0f / red[0];
#pragma unroll
  for (int i = 0; i < 4; ++i) p[t + i * 256] = (bf16)(v[i] * inv);
}

// tmp = attn_out * sigmoid(G)  (G already has gate bias from GEMM epilogue)
__global__ void mulgate_kernel(const float* o, const float* G, bf16* out, long n) {
  long i = (long)blockIdx.x * blockDim.x + threadIdx.x;
  if (i < n) out[i] = (bf16)(o[i] * sigmoidf_(G[i]));
}

// x = LN(src + attn) over D -> xF (f32) and xB (bf16)
__global__ void __launch_bounds__(256) ln1_kernel(const float* src, const float* attn,
                                                  const float* g, const float* bt,
                                                  float* xF, bf16* xB) {
  __shared__ float red[256];
  long tk = blockIdx.x; int t = threadIdx.x;
  float v[4]; float s = 0.f;
#pragma unroll
  for (int i = 0; i < 4; ++i) { long o = tk * DD + t + i * 256; v[i] = src[o] + attn[o]; s += v[i]; }
  red[t] = s; __syncthreads();
  for (int q = 128; q > 0; q >>= 1) { if (t < q) red[t] += red[t + q]; __syncthreads(); }
  float mean = red[0] * (1.0f / DD); __syncthreads();
  s = 0.f;
#pragma unroll
  for (int i = 0; i < 4; ++i) { float d = v[i] - mean; s += d * d; }
  red[t] = s; __syncthreads();
  for (int q = 128; q > 0; q >>= 1) { if (t < q) red[t] += red[t + q]; __syncthreads(); }
  float inv = rsqrtf(red[0] * (1.0f / DD) + 1e-5f);
#pragma unroll
  for (int i = 0; i < 4; ++i) {
    int c = t + i * 256; long o = tk * DD + c;
    float y = (v[i] - mean) * inv * g[c] + bt[c];
    xF[o] = y; xB[o] = (bf16)y;
  }
}

__global__ void zero_i32(int* p, int n) { int i = threadIdx.x; if (i < n) p[i] = 0; }

// MoE router: logits = x@gw + gb; chosen = max index of top-2 (softmax of top-2 sums to 1)
__global__ void route_kernel(const float* xF, const float* gw, const float* gb,
                             int* chosen, int* cnt, int* idx) {
  int t = blockIdx.x * blockDim.x + threadIdx.x;
  if (t >= NT) return;
  float l[EE];
#pragma unroll
  for (int e = 0; e < EE; ++e) l[e] = gb[e];
  const float* x = xF + (long)t * DD;
  for (int d = 0; d < DD; ++d) {
    float xv = x[d];
    const float* w = gw + (long)d * EE;
#pragma unroll
    for (int e = 0; e < EE; ++e) l[e] += xv * w[e];
  }
  float b1v = -3.0e38f; int i1 = 0;
#pragma unroll
  for (int e = 0; e < EE; ++e) if (l[e] > b1v) { b1v = l[e]; i1 = e; }
  float b2v = -3.0e38f; int i2 = 0;
#pragma unroll
  for (int e = 0; e < EE; ++e) if (e != i1 && l[e] > b2v) { b2v = l[e]; i2 = e; }
  int ch = (i1 > i2) ? i1 : i2;
  chosen[t] = ch;
  int pos = atomicAdd(&cnt[ch], 1);
  idx[ch * NT + pos] = t;
}

// h -> gelu(LN_e(h)) bf16, per token over F (expert-specific gamma/beta)
__global__ void __launch_bounds__(256) lngelu_kernel(const float* hF, const int* chosen,
                                                     const float* lg, const float* lb, bf16* hB) {
  __shared__ float red[256];
  long tk = blockIdx.x; int t = threadIdx.x;
  int e = chosen[tk];
  const float* g = lg + (long)e * FF; const float* bt = lb + (long)e * FF;
  float v[8]; float s = 0.f;
#pragma unroll
  for (int i = 0; i < 8; ++i) { v[i] = hF[tk * FF + t + i * 256]; s += v[i]; }
  red[t] = s; __syncthreads();
  for (int q = 128; q > 0; q >>= 1) { if (t < q) red[t] += red[t + q]; __syncthreads(); }
  float mean = red[0] * (1.0f / FF); __syncthreads();
  s = 0.f;
#pragma unroll
  for (int i = 0; i < 8; ++i) { float d = v[i] - mean; s += d * d; }
  red[t] = s; __syncthreads();
  for (int q = 128; q > 0; q >>= 1) { if (t < q) red[t] += red[t + q]; __syncthreads(); }
  float inv = rsqrtf(red[0] * (1.0f / FF) + 1e-5f);
#pragma unroll
  for (int i = 0; i < 8; ++i) {
    int c = t + i * 256;
    float y = (v[i] - mean) * inv * g[c] + bt[c];
    y = 0.5f * y * (1.0f + erff(y * 0.70710678118654752f));   // exact gelu
    hB[tk * FF + c] = (bf16)y;
  }
}

// out = LN(2*x + y*res_scale[chosen]) over D   (y already includes b2)
__global__ void __launch_bounds__(256) ln2_kernel(const float* xF, const float* yF,
                                                  const int* chosen, const float* rscale,
                                                  const float* g, const float* bt, float* out) {
  __shared__ float red[256];
  long tk = blockIdx.x; int t = threadIdx.x;
  float rs = rscale[chosen[tk]];
  float v[4]; float s = 0.f;
#pragma unroll
  for (int i = 0; i < 4; ++i) {
    long o = tk * DD + t + i * 256;
    v[i] = 2.0f * xF[o] + yF[o] * rs; s += v[i];
  }
  red[t] = s; __syncthreads();
  for (int q = 128; q > 0; q >>= 1) { if (t < q) red[t] += red[t + q]; __syncthreads(); }
  float mean = red[0] * (1.0f / DD); __syncthreads();
  s = 0.f;
#pragma unroll
  for (int i = 0; i < 4; ++i) { float d = v[i] - mean; s += d * d; }
  red[t] = s; __syncthreads();
  for (int q = 128; q > 0; q >>= 1) { if (t < q) red[t] += red[t + q]; __syncthreads(); }
  float inv = rsqrtf(red[0] * (1.0f / DD) + 1e-5f);
#pragma unroll
  for (int i = 0; i < 4; ++i) {
    int c = t + i * 256;
    out[tk * DD + c] = (v[i] - mean) * inv * g[c] + bt[c];
  }
}

// =======================================================================
// Host-side orchestration
// =======================================================================
extern "C" void kernel_launch(void* const* d_in, const int* in_sizes, int n_in,
                              void* d_out, int out_size, void* d_ws, size_t ws_size,
                              hipStream_t stream) {
  (void)in_sizes; (void)n_in; (void)out_size; (void)ws_size;
  const float* src        = (const float*)d_in[0];
  const float* q_w        = (const float*)d_in[1];
  const float* k_w        = (const float*)d_in[2];
  const float* v_w        = (const float*)d_in[3];
  const float* out_w      = (const float*)d_in[4];
  const float* gate_w     = (const float*)d_in[5];
  const float* gate_b     = (const float*)d_in[6];
  const float* scale_w    = (const float*)d_in[7];
  const float* n1_g       = (const float*)d_in[8];
  const float* n1_b       = (const float*)d_in[9];
  const float* n2_g       = (const float*)d_in[10];
  const float* n2_b       = (const float*)d_in[11];
  const float* moe_gate_w = (const float*)d_in[12];
  const float* moe_gate_b = (const float*)d_in[13];
  const float* w1         = (const float*)d_in[14];
  const float* b1         = (const float*)d_in[15];
  const float* ln_g       = (const float*)d_in[16];
  const float* ln_b       = (const float*)d_in[17];
  const float* w2         = (const float*)d_in[18];
  const float* b2         = (const float*)d_in[19];
  const float* res_scale  = (const float*)d_in[20];
  float* out = (float*)d_out;

  // ---- carve workspace (~190 MiB needed) ----
  char* base = (char*)d_ws; size_t off = 0;
  auto carve = [&](size_t bytes) -> void* {
    off = (off + 255) & ~(size_t)255;
    void* p = base + off; off += bytes; return p;
  };
  bf16* srcB = (bf16*)carve((size_t)NT * DD * 2);
  bf16* qwT  = (bf16*)carve((size_t)DD * DD * 2);
  bf16* kwT  = (bf16*)carve((size_t)DD * DD * 2);
  bf16* vwT  = (bf16*)carve((size_t)DD * DD * 2);
  bf16* owT  = (bf16*)carve((size_t)DD * DD * 2);
  bf16* gwT  = (bf16*)carve((size_t)DD * DD * 2);
  bf16* w1T  = (bf16*)carve((size_t)EE * DD * FF * 2);
  bf16* w2T  = (bf16*)carve((size_t)EE * DD * FF * 2);
  float* Qf  = (float*)carve((size_t)NT * DD * 4);   // later aliased as attn2F
  bf16* Qb   = (bf16*)carve((size_t)NT * DD * 2);
  bf16* Kb   = (bf16*)carve((size_t)NT * DD * 2);
  bf16* Vb   = (bf16*)carve((size_t)NT * DD * 2);
  bf16* Vt   = (bf16*)carve((size_t)NT * DD * 2);
  float* scl = (float*)carve((size_t)BB * HH * SS * 4);
  float* attnF = (float*)carve((size_t)NT * DD * 4);
  bf16*  attnB = (bf16*)carve((size_t)NT * DD * 2);
  float* xF  = (float*)carve((size_t)NT * DD * 4);
  bf16*  xB  = (bf16*)carve((size_t)NT * DD * 2);
  int* chosen = (int*)carve((size_t)NT * 4);
  int* cnt    = (int*)carve((size_t)EE * 4);
  int* idx    = (int*)carve((size_t)EE * NT * 4);
  char* big   = (char*)carve((size_t)BB * HH * SS * SS * 2);  // 64 MiB, phase-aliased
  bf16*  scoresB = (bf16*)big;                                  // attention phase
  float* Gf      = (float*)big;                                 // gate phase
  bf16*  tmpB    = (bf16*)(big + (size_t)NT * DD * 4);
  float* attn2F  = Qf;                                          // alias (Qf dead)
  float* hF      = (float*)big;                                 // MoE phase
  bf16*  hB      = (bf16*)(big + (size_t)NT * FF * 4);
  float* yF      = (float*)(big + (size_t)NT * FF * 4 + (size_t)NT * FF * 2);

  const dim3 blk16(16, 16);

  // ---- stage 0: bf16 conversions / weight transposes ----
  cvt_f32_bf16<<<(NT * DD) / 256, 256, 0, stream>>>(src, srcB, (long)NT * DD);
  tconv_kernel<<<dim3(DD/16, DD/16, 1), blk16, 0, stream>>>(q_w, qwT, DD, DD);
  tconv_kernel<<<dim3(DD/16, DD/16, 1), blk16, 0, stream>>>(k_w, kwT, DD, DD);
  tconv_kernel<<<dim3(DD/16, DD/16, 1), blk16, 0, stream>>>(v_w, vwT, DD, DD);
  tconv_kernel<<<dim3(DD/16, DD/16, 1), blk16, 0, stream>>>(out_w, owT, DD, DD);
  tconv_kernel<<<dim3(DD/16, DD/16, 1), blk16, 0, stream>>>(gate_w, gwT, DD, DD);
  tconv_kernel<<<dim3(FF/16, DD/16, EE), blk16, 0, stream>>>(w1, w1T, DD, FF);
  tconv_kernel<<<dim3(DD/16, FF/16, EE), blk16, 0, stream>>>(w2, w2T, FF, DD);

  auto G0 = []() { GemmParams p{}; p.zdiv = 1; p.rowidxStride = 0; return p; };

  // ---- QKV projections: [NT,D] = srcB @ W^T ----
  {
    GemmParams p = G0();
    p.A = srcB; p.lda = DD; p.ldb = DD; p.ldc = DD;
    p.M = NT; p.N = DD; p.K = DD;
    p.Bt = qwT; p.Cf = Qf; p.Cb = Qb;
    gemm_bf16_wmma<<<dim3(DD/64, NT/BM, 1), 128, 0, stream>>>(p);
    p.Bt = kwT; p.Cf = nullptr; p.Cb = Kb;
    gemm_bf16_wmma<<<dim3(DD/64, NT/BM, 1), 128, 0, stream>>>(p);
    p.Bt = vwT; p.Cb = Vb;
    gemm_bf16_wmma<<<dim3(DD/64, NT/BM, 1), 128, 0, stream>>>(p);
  }
  vt_kernel<<<(NT * DD) / 256, 256, 0, stream>>>(Vb, Vt);
  scale_kernel<<<(BB * HH * SS) / 256, 256, 0, stream>>>(Qf, scale_w, scl);

  // ---- scores[bh] = (Q_h @ K_h^T) * rowscale, bf16 out ----
  {
    GemmParams p = G0();
    p.zdiv = HH;
    p.A = Qb;  p.lda = DD; p.bsA1 = (long)SS * DD; p.bsA2 = HDD;
    p.Bt = Kb; p.ldb = DD; p.bsB1 = (long)SS * DD; p.bsB2 = HDD;
    p.Cb = scoresB; p.ldc = SS; p.bsC1 = (long)HH * SS * SS; p.bsC2 = (long)SS * SS;
    p.rowscale = scl; p.bsRS1 = (long)HH * SS; p.bsRS2 = SS;
    p.M = SS; p.N = SS; p.K = HDD;
    gemm_bf16_wmma<<<dim3(SS/64, SS/BM, BB * HH), 128, 0, stream>>>(p);
  }
  softmax_kernel<<<BB * HH * SS, 256, 0, stream>>>(scoresB);

  // ---- attn_out[bh] = P @ V_h  -> token-major [b,s,h,hd] (f32 + bf16) ----
  {
    GemmParams p = G0();
    p.zdiv = HH;
    p.A = scoresB; p.lda = SS; p.bsA1 = (long)HH * SS * SS; p.bsA2 = (long)SS * SS;
    p.Bt = Vt; p.ldb = SS; p.bsB1 = (long)HH * HDD * SS; p.bsB2 = (long)HDD * SS;
    p.Cf = attnF; p.Cb = attnB; p.ldc = DD; p.bsC1 = (long)SS * DD; p.bsC2 = HDD;
    p.M = SS; p.N = HDD; p.K = SS;
    gemm_bf16_wmma<<<dim3(1, SS/BM, BB * HH), 128, 0, stream>>>(p);
  }

  // ---- gate = sigmoid(attn_out @ gate_w + gate_b); attn2 = (attn_out*gate) @ out_w ----
  {
    GemmParams p = G0();
    p.A = attnB; p.lda = DD; p.Bt = gwT; p.ldb = DD;
    p.Cf = Gf; p.ldc = DD; p.bias = gate_b; p.bsBias = 0;
    p.M = NT; p.N = DD; p.K = DD;
    gemm_bf16_wmma<<<dim3(DD/64, NT/BM, 1), 128, 0, stream>>>(p);
  }
  mulgate_kernel<<<(NT * DD) / 256, 256, 0, stream>>>(attnF, Gf, tmpB, (long)NT * DD);
  {
    GemmParams p = G0();
    p.A = tmpB; p.lda = DD; p.Bt = owT; p.ldb = DD;
    p.Cf = attn2F; p.ldc = DD;
    p.M = NT; p.N = DD; p.K = DD;
    gemm_bf16_wmma<<<dim3(DD/64, NT/BM, 1), 128, 0, stream>>>(p);
  }

  // ---- x = LN(src + attn2) ----
  ln1_kernel<<<NT, 256, 0, stream>>>(src, attn2F, n1_g, n1_b, xF, xB);

  // ---- MoE routing ----
  zero_i32<<<1, 32, 0, stream>>>(cnt, EE);
  route_kernel<<<NT / 256, 256, 0, stream>>>(xF, moe_gate_w, moe_gate_b, chosen, cnt, idx);

  // ---- grouped GEMM 1: h = gather(x) @ W1[e]^T + b1[e] ----
  {
    GemmParams p = G0();
    p.A = xB; p.lda = DD;
    p.Bt = w1T; p.ldb = DD; p.bsB1 = (long)FF * DD;
    p.Cf = hF; p.ldc = FF;
    p.bias = b1; p.bsBias = FF;
    p.rowidx = idx; p.rowidxStride = NT; p.counts = cnt;
    p.M = NT; p.N = FF; p.K = DD;
    gemm_bf16_wmma<<<dim3(FF/64, NT/BM, EE), 128, 0, stream>>>(p);
  }
  lngelu_kernel<<<NT, 256, 0, stream>>>(hF, chosen, ln_g, ln_b, hB);

  // ---- grouped GEMM 2: y = gather(h) @ W2[e]^T + b2[e] ----
  {
    GemmParams p = G0();
    p.A = hB; p.lda = FF;
    p.Bt = w2T; p.ldb = FF; p.bsB1 = (long)DD * FF;
    p.Cf = yF; p.ldc = DD;
    p.bias = b2; p.bsBias = DD;
    p.rowidx = idx; p.rowidxStride = NT; p.counts = cnt;
    p.M = NT; p.N = DD; p.K = FF;
    gemm_bf16_wmma<<<dim3(DD/64, NT/BM, EE), 128, 0, stream>>>(p);
  }

  // ---- final: out = LN(2x + y*res_scale[chosen]) ----
  ln2_kernel<<<NT, 256, 0, stream>>>(xF, yF, chosen, res_scale, n2_g, n2_b, out);
}
